// Model_24335284699649
// MI455X (gfx1250) — matrix operand order repaired and checked by reading, not compiled, not run on hardware
//
#include <hip/hip_runtime.h>
#include <math.h>

// ---------------------------------------------------------------------------
// Model constants (from reference)
// ---------------------------------------------------------------------------
#define BB 4
#define NN 8192
#define XD 7
#define YD 4
#define CC 256
#define HH 8
#define GG 64
#define DD 32
#define LL 8
#define MM (BB * NN)      // 32768 tokens
#define BH (BB * HH)      // 32
#define EPSF 1e-5f

typedef _Float16 half_t;
typedef __attribute__((ext_vector_type(16))) _Float16 v16h;
typedef __attribute__((ext_vector_type(8)))  _Float16 v8h;
typedef __attribute__((ext_vector_type(8)))  float    v8f;
typedef __attribute__((ext_vector_type(4)))  int      v4i;

__device__ __forceinline__ float gelu_f(float x) {
    return 0.5f * x * (1.0f + erff(x * 0.70710678118654752f));
}

#define SHUF16(lo, hi) __builtin_shufflevector((lo), (hi), 0, 1, 2, 3, 4, 5, 6, 7, \
                                               8, 9, 10, 11, 12, 13, 14, 15)

// CDNA5 async global->LDS copy path (ASYNCcnt-tracked), guarded so the file
// still compiles on toolchains without the builtins.
// Builtin signature (from hipcc diagnostics):
//   void __builtin_amdgcn_global_load_async_to_lds_b128(
//        v4i addrspace(1)* src, v4i addrspace(3)* dst, imm offset, imm cpol)
#if __has_builtin(__builtin_amdgcn_global_load_async_to_lds_b128) && \
    __has_builtin(__builtin_amdgcn_s_wait_asynccnt)
#define USE_ASYNC_COPY 1
#define GAS(p) ((__attribute__((address_space(1))) v4i*)(p))
#define LAS(p) ((__attribute__((address_space(3))) v4i*)(p))
#else
#define USE_ASYNC_COPY 0
#endif

// ---------------------------------------------------------------------------
// f32 [L][K=256][N=256] -> f16 transposed per layer: dst[l][n][k]
// ---------------------------------------------------------------------------
__global__ __launch_bounds__(256) void convert_f16_T(const float* __restrict__ src,
                                                     half_t* __restrict__ dst) {
    const int l = blockIdx.y;
    const int i = blockIdx.x * 256 + threadIdx.x;   // 0..65535
    const int k = i >> 8, n = i & 255;
    dst[(size_t)l * CC * CC + n * CC + k] = (half_t)src[(size_t)l * CC * CC + k * CC + n];
}

__global__ __launch_bounds__(256) void zero_f32(float* __restrict__ p, int n) {
    int i = blockIdx.x * 256 + threadIdx.x;
    if (i < n) p[i] = 0.0f;
}

// ---------------------------------------------------------------------------
// Preprocess MLP: h = gelu(x @ w1 + b1) @ w2 + b2 + placeholder
// ---------------------------------------------------------------------------
__global__ __launch_bounds__(256) void preprocess_k(
    const float* __restrict__ x, const float* __restrict__ w1, const float* __restrict__ b1,
    const float* __restrict__ w2, const float* __restrict__ b2,
    const float* __restrict__ ph, float* __restrict__ hbuf) {
    const int m = blockIdx.x;
    const int t = threadIdx.x;
    __shared__ float xs[XD];
    __shared__ float us[2 * CC];
    if (t < XD) xs[t] = x[(size_t)m * XD + t];
    __syncthreads();
    for (int j = t; j < 2 * CC; j += 256) {
        float a = b1[j];
#pragma unroll
        for (int k = 0; k < XD; ++k) a += xs[k] * w1[k * (2 * CC) + j];
        us[j] = gelu_f(a);
    }
    __syncthreads();
    float a = b2[t];
    for (int k = 0; k < 2 * CC; ++k) a += us[k] * w2[k * CC + t];
    hbuf[(size_t)m * CC + t] = a + ph[t];
}

// ---------------------------------------------------------------------------
// LayerNorm over C=256, output f16. One block per token.
// ---------------------------------------------------------------------------
__global__ __launch_bounds__(256) void layernorm_f16_k(
    const float* __restrict__ hbuf, const float* __restrict__ w, const float* __restrict__ b,
    half_t* __restrict__ z16) {
    const int m = blockIdx.x;
    const int c = threadIdx.x;
    __shared__ float red[256];
    float x = hbuf[(size_t)m * CC + c];
    red[c] = x; __syncthreads();
    for (int off = 128; off > 0; off >>= 1) {
        if (c < off) red[c] += red[c + off];
        __syncthreads();
    }
    float mean = red[0] * (1.0f / CC);
    __syncthreads();
    float d = x - mean;
    red[c] = d * d; __syncthreads();
    for (int off = 128; off > 0; off >>= 1) {
        if (c < off) red[c] += red[c + off];
        __syncthreads();
    }
    float rstd = rsqrtf(red[0] * (1.0f / CC) + EPSF);
    z16[(size_t)m * CC + c] = (half_t)(d * rstd * w[c] + b[c]);
}

// ---------------------------------------------------------------------------
// Core WMMA GEMM: Y[M,256] = A16[M,256] @ W[256,256], W given TRANSPOSED
// (Wt[n][k]) so B fragments are contiguous-run LDS reads.
// Block: 256 thr (8 waves), tile 128(M) x 64(N); each wave computes 32x32.
// Async path: GLOBAL_LOAD_ASYNC_TO_LDS_B128 double-buffered pipeline.
//   EPI 0: out16 = (f16)(acc + bias)
//   EPI 1: out16 = (f16)gelu(acc + bias)
//   EPI 2: resid[m,n] += acc + bias      (in-place residual update, f32)
// ---------------------------------------------------------------------------
template <int EPI>
__global__ __launch_bounds__(256) void gemm_k256(
    const half_t* __restrict__ A, const half_t* __restrict__ Wt,
    const float* __restrict__ bias, float* __restrict__ resid,
    half_t* __restrict__ out16) {
    const int t = threadIdx.x;
    const int lane = t & 31;
    const int wave = t >> 5;
    const int wr = wave >> 1;        // 0..3  (32-row group)
    const int wc = wave & 1;         // 0..1  (32-col group)
    const int mBase = blockIdx.x * 128;
    const int nBase = blockIdx.y * 64;

    v8f acc[2][2] = {};

    const int lrow = t >> 2, lseg = (t & 3) * 8;    // loader: 32-wide K rows
    const int mfr = lane & 15;
    const int koff = (lane >> 4) * 8;               // unused; A also uses kh
    const int kh = (lane >> 4) * 16;                // B half-lane K offset
    const int ncl = lane & 15;

#if USE_ASYNC_COPY
    __shared__ __align__(16) half_t As[2][128 * 40];
    __shared__ __align__(16) half_t Bs[2][64 * 40];
    // 3 async 16B copies per lane per tile (2 A rows + 1 B row)
    auto issue = [&](int kb, int p) {
        __builtin_amdgcn_global_load_async_to_lds_b128(
            GAS(const_cast<half_t*>(A) + (size_t)(mBase + lrow) * CC + kb + lseg),
            LAS(&As[p][lrow * 40 + lseg]), 0, 0);
        __builtin_amdgcn_global_load_async_to_lds_b128(
            GAS(const_cast<half_t*>(A) + (size_t)(mBase + 64 + lrow) * CC + kb + lseg),
            LAS(&As[p][(lrow + 64) * 40 + lseg]), 0, 0);
        __builtin_amdgcn_global_load_async_to_lds_b128(
            GAS(const_cast<half_t*>(Wt) + (size_t)(nBase + lrow) * CC + kb + lseg),
            LAS(&Bs[p][lrow * 40 + lseg]), 0, 0);
    };
    issue(0, 0);
    int p = 0;
    for (int kb = 0; kb < 256; kb += 32) {
        if (kb < 224) {
            issue(kb + 32, p ^ 1);
            __builtin_amdgcn_s_wait_asynccnt(3);  // oldest tile's 3 ops retired
        } else {
            __builtin_amdgcn_s_wait_asynccnt(0);
        }
        __syncthreads();                          // tile p visible to all waves

        v16h afr[2], bfr[2];
#pragma unroll
        for (int i = 0; i < 2; ++i) {
            const half_t* ap = &As[p][(wr * 32 + i * 16 + mfr) * 40];
            afr[i] = SHUF16(*(const v8h*)(ap + kh), *(const v8h*)(ap + kh + 8));
            const half_t* bp = &Bs[p][(wc * 32 + i * 16 + ncl) * 40 + kh];
            bfr[i] = SHUF16(*(const v8h*)(bp), *(const v8h*)(bp + 8));
        }
#pragma unroll
        for (int i = 0; i < 2; ++i)
#pragma unroll
            for (int j = 0; j < 2; ++j)
                acc[i][j] = __builtin_amdgcn_wmma_f32_16x16x32_f16(
                    false, afr[i], false, bfr[j], (short)0, acc[i][j], false, false);
        __syncthreads();                          // done reading buf p
        p ^= 1;
    }
#else
    __shared__ __align__(16) half_t As[128 * 40];   // [m][k] natural
    __shared__ __align__(16) half_t Bs[64 * 40];    // [n][k] (W transposed)
    for (int kb = 0; kb < 256; kb += 32) {
        *(v8h*)(As + lrow * 40 + lseg) =
            *(const v8h*)(A + (size_t)(mBase + lrow) * CC + kb + lseg);
        *(v8h*)(As + (lrow + 64) * 40 + lseg) =
            *(const v8h*)(A + (size_t)(mBase + 64 + lrow) * CC + kb + lseg);
        *(v8h*)(Bs + lrow * 40 + lseg) =
            *(const v8h*)(Wt + (size_t)(nBase + lrow) * CC + kb + lseg);
        if (kb + 32 < 256) {
            __builtin_prefetch(A + (size_t)(mBase + lrow) * CC + kb + 32 + lseg, 0, 0);
            __builtin_prefetch(Wt + (size_t)(nBase + lrow) * CC + kb + 32 + lseg, 0, 0);
        }
        __syncthreads();
        v16h afr[2], bfr[2];
#pragma unroll
        for (int i = 0; i < 2; ++i) {
            const half_t* ap = As + (wr * 32 + i * 16 + mfr) * 40;
            afr[i] = SHUF16(*(const v8h*)(ap + kh), *(const v8h*)(ap + kh + 8));
            const half_t* bp = Bs + (wc * 32 + i * 16 + ncl) * 40 + kh;
            bfr[i] = SHUF16(*(const v8h*)(bp), *(const v8h*)(bp + 8));
        }
#pragma unroll
        for (int i = 0; i < 2; ++i)
#pragma unroll
            for (int j = 0; j < 2; ++j)
                acc[i][j] = __builtin_amdgcn_wmma_f32_16x16x32_f16(
                    false, afr[i], false, bfr[j], (short)0, acc[i][j], false, false);
        __syncthreads();
    }
#endif

#pragma unroll
    for (int i = 0; i < 2; ++i) {
#pragma unroll
        for (int j = 0; j < 2; ++j) {
#pragma unroll
            for (int v = 0; v < 8; ++v) {
                const int row = mBase + wr * 32 + i * 16 + ((lane < 16) ? v : v + 8);
                const int col = nBase + wc * 32 + j * 16 + ncl;
                float xv = acc[i][j][v] + bias[col];
                if (EPI == 0) {
                    out16[(size_t)row * CC + col] = (half_t)xv;
                } else if (EPI == 1) {
                    out16[(size_t)row * CC + col] = (half_t)gelu_f(xv);
                } else {
                    resid[(size_t)row * CC + col] += xv;
                }
            }
        }
    }
}

// ---------------------------------------------------------------------------
// Slice softmax: logits[b,h,n,g] = (xm_head[32] @ sl_w[32,64] + sb)/temp[h]
// ---------------------------------------------------------------------------
__global__ __launch_bounds__(64) void slice_softmax_k(
    const half_t* __restrict__ xm16, const float* __restrict__ slw,
    const float* __restrict__ slb, const float* __restrict__ temp,
    half_t* __restrict__ sw16) {
    const int idx = blockIdx.x;          // 0 .. BH*N-1
    const int n = idx & (NN - 1);
    const int bh = idx >> 13;            // N == 8192 == 2^13
    const int h = bh & (HH - 1);
    const int b = bh >> 3;
    const int g = threadIdx.x;
    __shared__ float xs[DD];
    __shared__ float ls[GG];
    __shared__ float ps[GG];
    if (g < DD) xs[g] = (float)xm16[(size_t)(b * NN + n) * CC + h * DD + g];
    __syncthreads();
    float l = 0.0f;
#pragma unroll
    for (int d = 0; d < DD; ++d) l += xs[d] * slw[d * GG + g];
    l = (l + slb[g]) / temp[h];
    ls[g] = l; __syncthreads();
    float mx = -1e30f;
    for (int j = 0; j < GG; ++j) mx = fmaxf(mx, ls[j]);
    float p = __expf(l - mx);
    ps[g] = p; __syncthreads();
    float s = 0.0f;
    for (int j = 0; j < GG; ++j) s += ps[j];
    sw16[(size_t)bh * NN * GG + (size_t)n * GG + g] = (half_t)(p / s);
}

// ---------------------------------------------------------------------------
// snorm[bh,g] = sum_n sweights[bh,n,g]
// ---------------------------------------------------------------------------
__global__ __launch_bounds__(256) void snorm_k(const half_t* __restrict__ sw16,
                                               float* __restrict__ snorm) {
    const int blk = blockIdx.x;
    const int bh = blk >> 6;
    const int g = blk & 63;
    const half_t* p = sw16 + (size_t)bh * NN * GG + g;
    float s = 0.0f;
    for (int n = threadIdx.x; n < NN; n += 256) s += (float)p[(size_t)n * GG];
    __shared__ float red[256];
    red[threadIdx.x] = s; __syncthreads();
    for (int off = 128; off > 0; off >>= 1) {
        if (threadIdx.x < off) red[threadIdx.x] += red[threadIdx.x + off];
        __syncthreads();
    }
    if (threadIdx.x == 0) snorm[bh * GG + g] = red[0];
}

// ---------------------------------------------------------------------------
// stoken[bh,g,d] = sum_n fx[b,n,h,d]*sw[bh,n,g]  (WMMA, split-K, atomics).
// Both LDS tiles are written TRANSPOSED (scatter b16 stores) so that all
// fragment reads are contiguous b128 runs:
//   Asw_t[g][n]  (A = sw^T, m=g, k=n)      Bfx_t[d][n]  (B col-major, col=d)
// ---------------------------------------------------------------------------
__global__ __launch_bounds__(256) void stoken_gemm_k(
    const half_t* __restrict__ sw16, const half_t* __restrict__ fx16,
    float* __restrict__ stoken) {
    __shared__ __align__(16) half_t Asw[64 * 40];   // [g][n]
    __shared__ __align__(16) half_t Bfx[32 * 40];   // [d][n]
    const int bh = blockIdx.x;
    const int b = bh >> 3, h = bh & 7;
    const int ks = blockIdx.y;                      // 0..15, 512 tokens each
    const int t = threadIdx.x;
    const int lane = t & 31;
    const int wave = t >> 5;
    const int gr = wave >> 1;                       // 0..3
    const int dc = wave & 1;                        // 0..1
    v8f acc = {};

    const int mfr = lane & 15;
    const int koff = (lane >> 4) * 8;
    const int kh = (lane >> 4) * 16;
    const int ncl = lane & 15;

    const int srow = t >> 3, sseg = (t & 7) * 8;    // sw tile: 32 n x 64 g
    for (int kk = 0; kk < 512; kk += 32) {
        const int n0 = ks * 512 + kk;
        {
            v8h val = *(const v8h*)(sw16 + (size_t)bh * NN * GG +
                                    (size_t)(n0 + srow) * GG + sseg);
#pragma unroll
            for (int j = 0; j < 8; ++j) Asw[(sseg + j) * 40 + srow] = val[j];
        }
        if (t < 128) {
            const int frow = t >> 2, fseg = (t & 3) * 8;   // fx tile: 32 n x 32 d
            v8h val = *(const v8h*)(fx16 + (size_t)(b * NN + n0 + frow) * CC +
                                    h * DD + fseg);
#pragma unroll
            for (int j = 0; j < 8; ++j) Bfx[(fseg + j) * 40 + frow] = val[j];
        }
        __syncthreads();

        const half_t* ap = Asw + (gr * 16 + mfr) * 40;
        v16h af = SHUF16(*(const v8h*)(ap + kh), *(const v8h*)(ap + kh + 8));
        const half_t* bp = Bfx + (dc * 16 + ncl) * 40 + kh;
        v16h bf = SHUF16(*(const v8h*)(bp), *(const v8h*)(bp + 8));
        acc = __builtin_amdgcn_wmma_f32_16x16x32_f16(false, af, false, bf,
                                                     (short)0, acc, false, false);
        __syncthreads();
    }
#pragma unroll
    for (int v = 0; v < 8; ++v) {
        const int g = gr * 16 + ((lane < 16) ? v : v + 8);
        const int d = dc * 16 + ncl;
        atomicAdd(&stoken[(size_t)bh * GG * DD + g * DD + d], acc[v]);
    }
}

// ---------------------------------------------------------------------------
// Slice-token attention among G=64 tokens of dim D=32. One block per (b,h).
// Writes ot TRANSPOSED: otT[bh][d][g]  (for contiguous deslice B fragments).
// ---------------------------------------------------------------------------
__global__ __launch_bounds__(64) void slice_attn_k(
    const float* __restrict__ stoken, const float* __restrict__ snorm,
    const float* __restrict__ qw, const float* __restrict__ kw,
    const float* __restrict__ vw, half_t* __restrict__ otT16) {
    const int bh = blockIdx.x;
    const int g = threadIdx.x;
    __shared__ float st[GG][DD + 1];
    __shared__ float Km[GG][DD + 1];
    __shared__ float Vm[GG][DD + 1];
    __shared__ float Wq[DD * DD], Wk[DD * DD], Wv[DD * DD];
    __shared__ float P[GG][GG + 8];
    for (int i = g; i < DD * DD; i += GG) { Wq[i] = qw[i]; Wk[i] = kw[i]; Wv[i] = vw[i]; }
    const float inv = 1.0f / (snorm[bh * GG + g] + EPSF);
#pragma unroll
    for (int d = 0; d < DD; ++d)
        st[g][d] = stoken[(size_t)bh * GG * DD + g * DD + d] * inv;
    __syncthreads();
    float q[DD];
#pragma unroll
    for (int j = 0; j < DD; ++j) {
        float aq = 0.0f, ak = 0.0f, av = 0.0f;
#pragma unroll
        for (int d = 0; d < DD; ++d) {
            float s = st[g][d];
            aq += s * Wq[d * DD + j];
            ak += s * Wk[d * DD + j];
            av += s * Wv[d * DD + j];
        }
        q[j] = aq; Km[g][j] = ak; Vm[g][j] = av;
    }
    __syncthreads();
    const float scale = 0.17677669529663687f;  // 32^-0.5
    float mx = -1e30f;
    for (int kk = 0; kk < GG; ++kk) {
        float a = 0.0f;
#pragma unroll
        for (int d = 0; d < DD; ++d) a += q[d] * Km[kk][d];
        a *= scale;
        P[g][kk] = a;
        mx = fmaxf(mx, a);
    }
    float s = 0.0f;
    for (int kk = 0; kk < GG; ++kk) {
        float e = __expf(P[g][kk] - mx);
        P[g][kk] = e; s += e;
    }
    const float isr = 1.0f / s;
#pragma unroll
    for (int d = 0; d < DD; ++d) {
        float o = 0.0f;
        for (int kk = 0; kk < GG; ++kk) o += P[g][kk] * Vm[kk][d];
        otT16[(size_t)bh * GG * DD + d * GG + g] = (half_t)(o * isr);  // [d][g]
    }
}

// ---------------------------------------------------------------------------
// Deslice: ox[b,n,h,d] = sum_g sw[bh,n,g]*ot[bh,g,d]  — WMMA GEMM, K=64.
// A = sweights[n][g] (row-major in K) ; B from otT[d][g] (col-major).
// Tiles are direct copies -> async staging when available.
// ---------------------------------------------------------------------------
__global__ __launch_bounds__(256) void deslice_gemm_k(
    const half_t* __restrict__ sw16, const half_t* __restrict__ otT16,
    half_t* __restrict__ ox16) {
    __shared__ __align__(16) half_t Asw[64 * 72];   // [token][g]
    __shared__ __align__(16) half_t Bot[32 * 72];   // [d][g]
    const int bh = blockIdx.x;
    const int b = bh >> 3, h = bh & 7;
    const int nt = blockIdx.y;
    const int t = threadIdx.x;
    const int lane = t & 31;
    const int wave = t >> 5;
    const int tr = wave >> 1;
    const int dc = wave & 1;

    const int ar0 = t >> 3, aseg = (t & 7) * 8;     // A slots 0..255 and 256..511
    const int ar1 = (t + 256) >> 3;
    const int br = t >> 3, bseg = (t & 7) * 8;      // 32x64 B tile

#if USE_ASYNC_COPY
    __builtin_amdgcn_global_load_async_to_lds_b128(
        GAS(const_cast<half_t*>(sw16) + (size_t)bh * NN * GG +
            (size_t)(nt * 64 + ar0) * GG + aseg),
        LAS(&Asw[ar0 * 72 + aseg]), 0, 0);
    __builtin_amdgcn_global_load_async_to_lds_b128(
        GAS(const_cast<half_t*>(sw16) + (size_t)bh * NN * GG +
            (size_t)(nt * 64 + ar1) * GG + aseg),
        LAS(&Asw[ar1 * 72 + aseg]), 0, 0);
    __builtin_amdgcn_global_load_async_to_lds_b128(
        GAS(const_cast<half_t*>(otT16) + (size_t)bh * GG * DD + br * GG + bseg),
        LAS(&Bot[br * 72 + bseg]), 0, 0);
    __builtin_amdgcn_s_wait_asynccnt(0);
#else
    *(v8h*)(Asw + ar0 * 72 + aseg) =
        *(const v8h*)(sw16 + (size_t)bh * NN * GG + (size_t)(nt * 64 + ar0) * GG + aseg);
    *(v8h*)(Asw + ar1 * 72 + aseg) =
        *(const v8h*)(sw16 + (size_t)bh * NN * GG + (size_t)(nt * 64 + ar1) * GG + aseg);
    *(v8h*)(Bot + br * 72 + bseg) =
        *(const v8h*)(otT16 + (size_t)bh * GG * DD + br * GG + bseg);
#endif
    __syncthreads();

    const int mfr = lane & 15;
    const int koff = (lane >> 4) * 8;
    const int kh = (lane >> 4) * 16;
    const int ncl = lane & 15;

    v8f acc = {};
#pragma unroll
    for (int kb = 0; kb < 64; kb += 32) {
        const half_t* ap = Asw + (tr * 16 + mfr) * 72 + kb;
        v16h af = SHUF16(*(const v8h*)(ap + kh), *(const v8h*)(ap + kh + 8));
        const half_t* bp = Bot + (dc * 16 + ncl) * 72 + kb + kh;
        v16h bf = SHUF16(*(const v8h*)(bp), *(const v8h*)(bp + 8));
        acc = __builtin_amdgcn_wmma_f32_16x16x32_f16(false, af, false, bf,
                                                     (short)0, acc, false, false);
    }
#pragma unroll
    for (int v = 0; v < 8; ++v) {
        const int tok = nt * 64 + tr * 16 + ((lane < 16) ? v : v + 8);
        const int d = dc * 16 + ncl;
        ox16[(size_t)(b * NN + tok) * CC + h * DD + d] = (half_t)acc[v];
    }
}

// ---------------------------------------------------------------------------
// Final LN + head: out[m, 0..3] = LN(h) @ head_w + head_b
// ---------------------------------------------------------------------------
__global__ __launch_bounds__(256) void final_head_k(
    const float* __restrict__ hbuf, const float* __restrict__ lw, const float* __restrict__ lb,
    const float* __restrict__ hw, const float* __restrict__ hb, float* __restrict__ out) {
    const int m = blockIdx.x;
    const int c = threadIdx.x;
    __shared__ float red[256];
    __shared__ float zs[256];
    float x = hbuf[(size_t)m * CC + c];
    red[c] = x; __syncthreads();
    for (int off = 128; off > 0; off >>= 1) {
        if (c < off) red[c] += red[c + off];
        __syncthreads();
    }
    float mean = red[0] * (1.0f / CC);
    __syncthreads();
    float d = x - mean;
    red[c] = d * d; __syncthreads();
    for (int off = 128; off > 0; off >>= 1) {
        if (c < off) red[c] += red[c + off];
        __syncthreads();
    }
    float rstd = rsqrtf(red[0] * (1.0f / CC) + EPSF);
    zs[c] = d * rstd * lw[c] + lb[c];
    __syncthreads();
    if (c < YD) {
        float a = hb[c];
        for (int k = 0; k < CC; ++k) a += zs[k] * hw[k * YD + c];
        out[(size_t)m * YD + c] = a;
    }
}

// ---------------------------------------------------------------------------
// Host orchestration
// ---------------------------------------------------------------------------
extern "C" void kernel_launch(void* const* d_in, const int* in_sizes, int n_in,
                              void* d_out, int out_size, void* d_ws, size_t ws_size,
                              hipStream_t stream) {
    const float* x      = (const float*)d_in[0];
    const float* pre_w1 = (const float*)d_in[1];
    const float* pre_b1 = (const float*)d_in[2];
    const float* pre_w2 = (const float*)d_in[3];
    const float* pre_b2 = (const float*)d_in[4];
    const float* ph     = (const float*)d_in[5];
    const float* ln1_w  = (const float*)d_in[6];
    const float* ln1_b  = (const float*)d_in[7];
    const float* fx_w   = (const float*)d_in[8];
    const float* fx_b   = (const float*)d_in[9];
    const float* xp_w   = (const float*)d_in[10];
    const float* xp_b   = (const float*)d_in[11];
    const float* sl_w   = (const float*)d_in[12];
    const float* sl_b   = (const float*)d_in[13];
    const float* temp   = (const float*)d_in[14];
    const float* q_w    = (const float*)d_in[15];
    const float* k_w    = (const float*)d_in[16];
    const float* v_w    = (const float*)d_in[17];
    const float* out_w  = (const float*)d_in[18];
    const float* out_b  = (const float*)d_in[19];
    const float* ln2_w  = (const float*)d_in[20];
    const float* ln2_b  = (const float*)d_in[21];
    const float* m_w1   = (const float*)d_in[22];
    const float* m_b1   = (const float*)d_in[23];
    const float* m_w2   = (const float*)d_in[24];
    const float* m_b2   = (const float*)d_in[25];
    const float* lnf_w  = (const float*)d_in[26];
    const float* lnf_b  = (const float*)d_in[27];
    const float* head_w = (const float*)d_in[28];
    const float* head_b = (const float*)d_in[29];

    // ---- workspace carve (~123 MB) ----
    char* ws = (char*)d_ws;
    auto carve = [&](size_t bytes) {
        void* p = (void*)ws;
        ws += (bytes + 255) & ~(size_t)255;
        return p;
    };
    float*  hbuf   = (float*)carve((size_t)MM * CC * 4);          // residual stream
    half_t* a16    = (half_t*)carve((size_t)MM * CC * 2);         // z / ox / z2
    half_t* p16    = (half_t*)carve((size_t)MM * CC * 2);         // fx
    half_t* q16    = (half_t*)carve((size_t)MM * CC * 2);         // xm / gelu-mid
    half_t* sw16   = (half_t*)carve((size_t)BH * NN * GG * 2);    // slice weights
    half_t* wf_fx  = (half_t*)carve((size_t)LL * CC * CC * 2);    // transposed f16 weights
    half_t* wf_xp  = (half_t*)carve((size_t)LL * CC * CC * 2);
    half_t* wf_out = (half_t*)carve((size_t)LL * CC * CC * 2);
    half_t* wf_m1  = (half_t*)carve((size_t)LL * CC * CC * 2);
    half_t* wf_m2  = (half_t*)carve((size_t)LL * CC * CC * 2);
    float*  stok   = (float*)carve((size_t)BH * GG * DD * 4);
    float*  snorm  = (float*)carve((size_t)BH * GG * 4);
    half_t* otT16  = (half_t*)carve((size_t)BH * GG * DD * 2);

    const dim3 convGrid(CC * CC / 256, LL);
    convert_f16_T<<<convGrid, 256, 0, stream>>>(fx_w,  wf_fx);
    convert_f16_T<<<convGrid, 256, 0, stream>>>(xp_w,  wf_xp);
    convert_f16_T<<<convGrid, 256, 0, stream>>>(out_w, wf_out);
    convert_f16_T<<<convGrid, 256, 0, stream>>>(m_w1,  wf_m1);
    convert_f16_T<<<convGrid, 256, 0, stream>>>(m_w2,  wf_m2);

    preprocess_k<<<MM, 256, 0, stream>>>(x, pre_w1, pre_b1, pre_w2, pre_b2, ph, hbuf);

    const dim3 gemmGrid(MM / 128, CC / 64);
    for (int l = 0; l < LL; ++l) {
        layernorm_f16_k<<<MM, 256, 0, stream>>>(hbuf, ln1_w + l * CC, ln1_b + l * CC, a16);
        gemm_k256<0><<<gemmGrid, 256, 0, stream>>>(a16, wf_fx + (size_t)l * CC * CC,
                                                   fx_b + l * CC, nullptr, p16);
        gemm_k256<0><<<gemmGrid, 256, 0, stream>>>(a16, wf_xp + (size_t)l * CC * CC,
                                                   xp_b + l * CC, nullptr, q16);
        slice_softmax_k<<<BH * NN, 64, 0, stream>>>(q16, sl_w + (size_t)l * DD * GG,
                                                    sl_b + l * GG, temp + l * HH, sw16);
        zero_f32<<<(BH * GG * DD + 255) / 256, 256, 0, stream>>>(stok, BH * GG * DD);
        stoken_gemm_k<<<dim3(BH, 16), 256, 0, stream>>>(sw16, p16, stok);
        snorm_k<<<BH * GG, 256, 0, stream>>>(sw16, snorm);
        slice_attn_k<<<BH, 64, 0, stream>>>(stok, snorm, q_w + (size_t)l * DD * DD,
                                            k_w + (size_t)l * DD * DD,
                                            v_w + (size_t)l * DD * DD, otT16);
        deslice_gemm_k<<<dim3(BH, NN / 64), 256, 0, stream>>>(sw16, otT16, a16);
        gemm_k256<2><<<gemmGrid, 256, 0, stream>>>(a16, wf_out + (size_t)l * CC * CC,
                                                   out_b + l * CC, hbuf, nullptr);
        layernorm_f16_k<<<MM, 256, 0, stream>>>(hbuf, ln2_w + l * CC, ln2_b + l * CC, a16);
        gemm_k256<1><<<gemmGrid, 256, 0, stream>>>(a16, wf_m1 + (size_t)l * CC * CC,
                                                   m_b1 + l * CC, nullptr, q16);
        gemm_k256<2><<<gemmGrid, 256, 0, stream>>>(q16, wf_m2 + (size_t)l * CC * CC,
                                                   m_b2 + l * CC, hbuf, nullptr);
    }
    final_head_k<<<MM, 256, 0, stream>>>(hbuf, lnf_w, lnf_b, head_w, head_b, (float*)d_out);
}
